// EquiRNBlock1_1322849927404
// MI455X (gfx1250) — matrix operand order repaired and checked by reading, not compiled, not run on hardware
//
#include <hip/hip_runtime.h>
#include <hip/hip_bf16.h>

typedef __attribute__((ext_vector_type(16))) _Float16 v16h;
typedef __attribute__((ext_vector_type(8)))  _Float16 v8h;
typedef __attribute__((ext_vector_type(8)))  float    v8f;

#define Bsz 4
#define Nsz 512
#define Msz 512
#define Dsz 4
#define Hsz 32
#define Lsz 128

__device__ __forceinline__ v8f wmma16x16x32(v16h a, v16h b, v8f c) {
  // D(16x16 f32) = A(16x32 f16) * B(32x16 f16) + C
  return __builtin_amdgcn_wmma_f32_16x16x32_f16(false, a, false, b, (short)0, c,
                                                false, false);
}

__device__ __forceinline__ v16h cat16(v8h lo, v8h hi) {
  return __builtin_shufflevector(lo, hi, 0, 1, 2, 3, 4, 5, 6, 7, 8, 9, 10, 11,
                                 12, 13, 14, 15);
}

// WMMA B fragment (K=32 x N=16) taken from an LDS matrix stored row-major
// as [out_features][32 in_features] f16, selecting out columns g0..g0+15.
// Layout (contiguous k order used here, same as load_afrag): lane<16 -> col N=lane, K=0..15; lane>=16 -> col
// N=lane-16, K=16..31 (16 consecutive halves each).
__device__ __forceinline__ v16h load_bfrag(const _Float16* W, int g0, int lane) {
  const int g  = g0 + (lane & 15);
  const int ko = (lane < 16) ? 0 : 16;
  const _Float16* p = W + g * 32 + ko;
  v8h lo = *(const v8h*)(p);
  v8h hi = *(const v8h*)(p + 8);
  return cat16(lo, hi);
}

// WMMA A fragment (M=16 x K=32) from LDS row-major [16 points][32 feats] f16.
// Layout: lane<16 -> row M=lane, K=0..15;
//         lane>=16 -> row M=lane-16, K=16..31.
__device__ __forceinline__ v16h load_afrag(const _Float16* Abuf, int lane) {
  const int ko = (lane < 16) ? 0 : 16;
  const _Float16* p = Abuf + (lane & 15) * 32 + ko;
  v8h lo = *(const v8h*)(p);       // K = ko .. ko+7
  v8h hi = *(const v8h*)(p + 8);   // K = ko+8 .. ko+15
  return cat16(lo, hi);
}

struct __align__(64) WaveScratch {
  _Float16 bufA[16][32];  // f16 activation staging (h1 / e1)
  _Float16 bufB[16][32];  // f16 activation staging (e2)
  float    zbuf[16];      // per-m z accumulator
};

__global__ __launch_bounds__(256) void equirn_kernel(
    const float* __restrict__ Xg,    const float* __restrict__ Yg,
    const float* __restrict__ eqw1,  const float* __restrict__ eqb1g,
    const float* __restrict__ eqw2,  const float* __restrict__ eqb2g,
    const float* __restrict__ eqw3,  const float* __restrict__ eqb3g,
    const float* __restrict__ encw1, const float* __restrict__ encb1g,
    const float* __restrict__ encw2, const float* __restrict__ encb2g,
    const float* __restrict__ encw3, const float* __restrict__ encb3g,
    float* __restrict__ out) {
  __shared__ __align__(64) _Float16 W2eq[32 * 32];
  __shared__ __align__(64) _Float16 W2en[32 * 32];
  __shared__ __align__(64) _Float16 W3en[128 * 32];
  __shared__ float w1y[32], w1x[32], b1e[32], b2e[32], w3e[32];
  __shared__ float w1n[32], b1n[32], b2n[32], b3n[128];
  __shared__ WaveScratch scr[8];
  __shared__ float red[16][128];

  const int tid  = threadIdx.x;
  const int wave = tid >> 5;
  const int lane = tid & 31;
  const int l15  = lane & 15;
  const int hi   = lane >> 4;  // 0 or 1 (lane half)

  // ---- cooperative weight preload into LDS (f16 for WMMA B operands) ----
  for (int i = tid; i < 1024; i += 256) {
    W2eq[i] = (_Float16)eqw2[i];
    W2en[i] = (_Float16)encw2[i];
  }
  for (int i = tid; i < 4096; i += 256) W3en[i] = (_Float16)encw3[i];
  if (tid < 32) {
    w1y[tid] = eqw1[2 * tid + 0];  // multiplies Y (pairs[...,0])
    w1x[tid] = eqw1[2 * tid + 1];  // multiplies X (pairs[...,1])
    b1e[tid] = eqb1g[tid];
    b2e[tid] = eqb2g[tid];
    w3e[tid] = eqw3[tid];
    w1n[tid] = encw1[tid];
    b1n[tid] = encb1g[tid];
    b2n[tid] = encb2g[tid];
  }
  if (tid < 128) b3n[tid] = encb3g[tid];
  __syncthreads();

  const int wg = blockIdx.x;      // wg = b*N + n
  const int b  = wg >> 9;         // N = 512
  const int n  = wg & (Nsz - 1);
  const float Xv    = Xg[(b * Nsz + n) * Dsz + (l15 & 3)];  // d = point%4
  const float eqb3v = eqb3g[0];

  // Loop-invariant weight B-fragments held in VGPRs.
  v16h bEq0 = load_bfrag(W2eq, 0, lane);
  v16h bEq1 = load_bfrag(W2eq, 16, lane);
  v16h bEn0 = load_bfrag(W2en, 0, lane);
  v16h bEn1 = load_bfrag(W2en, 16, lane);
  v16h bW3[8];
#pragma unroll
  for (int fg = 0; fg < 8; ++fg) bW3[fg] = load_bfrag(W3en, fg * 16, lane);

  float rmax[8];
#pragma unroll
  for (int fg = 0; fg < 8; ++fg) rmax[fg] = -3.4e38f;

  WaveScratch& S = scr[wave];
  const int hb  = hi ? 16 : 0;  // feature-half this lane computes in layer1
  const int pr0 = hi ? 8 : 0;   // point-row base this lane holds in C frags
  const float w3lo = w3e[l15];
  const float w3hi = w3e[l15 + 16];
  const float b2lo = b2e[l15];
  const float b2hi = b2e[l15 + 16];
  const float* Yb = Yg + b * Msz * Dsz;

  // Each wave owns chunks of 16 m-values; all staging LDS is wave-private,
  // so no workgroup barrier is needed in this loop (same-wave DS is in-order).
  for (int chunk = wave; chunk < Msz / 16; chunk += 8) {
    const int mbase = chunk * 16;
    if (lane < 16) S.zbuf[lane] = (float)Dsz * eqb3v;  // z = sum_d(dot)+D*b3

    // ---- eq MLP over 64 points (16 m x 4 d), as 4 WMMA tiles of 16 ----
    // Point index p in tile s: global p = s*16+p = m_local*4 + d.
#pragma unroll
    for (int s = 0; s < 4; ++s) {
      const float Yv = Yb[mbase * Dsz + s * 16 + l15];
      // layer1: 2 -> 32 (VALU), write f16 tile [point][feat]
#pragma unroll
      for (int i = 0; i < 16; i += 2) {
        const int h0 = hb + i;
        float a0 = fmaxf(fmaf(w1y[h0],     Yv, fmaf(w1x[h0],     Xv, b1e[h0])),     0.f);
        float a1 = fmaxf(fmaf(w1y[h0 + 1], Yv, fmaf(w1x[h0 + 1], Xv, b1e[h0 + 1])), 0.f);
        union { _Float16 h[2]; unsigned u; } pk;
        pk.h[0] = (_Float16)a0;
        pk.h[1] = (_Float16)a1;
        *(unsigned*)&S.bufA[l15][h0] = pk.u;
      }
      // layer2: 32 -> 32 via two WMMAs
      v16h A  = load_afrag(&S.bufA[0][0], lane);
      v8f  c0 = {};
      v8f  c1 = {};
      c0 = wmma16x16x32(A, bEq0, c0);
      c1 = wmma16x16x32(A, bEq1, c1);
      // layer3 (32 -> 1) folded directly onto the C fragments:
      // lane holds feature cols g=l15 and l15+16 for point-rows pr0..pr0+7.
      // t_r = w3[g]*relu(c0[r]+b2[g]) + w3[g+16]*relu(c1[r]+b2[g+16]).
      // Rows r=0..3 -> m-offset (hi*2), rows 4..7 -> m-offset (hi*2+1).
      float za = 0.f, zb = 0.f;
#pragma unroll
      for (int r = 0; r < 8; ++r) {
        float h20 = fmaxf(c0[r] + b2lo, 0.f);
        float h21 = fmaxf(c1[r] + b2hi, 0.f);
        float t   = fmaf(w3lo, h20, w3hi * h21);
        if (r < 4) za += t; else zb += t;
      }
      // Sum over the 16 feature-lanes of this half (masks stay within half).
      za += __shfl_xor(za, 1, 32);
      za += __shfl_xor(za, 2, 32);
      za += __shfl_xor(za, 4, 32);
      za += __shfl_xor(za, 8, 32);
      zb += __shfl_xor(zb, 1, 32);
      zb += __shfl_xor(zb, 2, 32);
      zb += __shfl_xor(zb, 4, 32);
      zb += __shfl_xor(zb, 8, 32);
      if (l15 == 0) {  // one lane per half; distinct addresses (hi differs)
        S.zbuf[s * 4 + hi * 2 + 0] += za;
        S.zbuf[s * 4 + hi * 2 + 1] += zb;
      }
    }

    // ---- enc MLP over the 16 m's ----
    {  // layer1: 1 -> 32 (VALU)
      const float zm = S.zbuf[l15];
#pragma unroll
      for (int i = 0; i < 16; i += 2) {
        const int h0 = hb + i;
        float a0 = fmaxf(fmaf(w1n[h0],     zm, b1n[h0]),     0.f);
        float a1 = fmaxf(fmaf(w1n[h0 + 1], zm, b1n[h0 + 1]), 0.f);
        union { _Float16 h[2]; unsigned u; } pk;
        pk.h[0] = (_Float16)a0;
        pk.h[1] = (_Float16)a1;
        *(unsigned*)&S.bufA[l15][h0] = pk.u;
      }
    }
    // layer2: 32 -> 32
    v16h A1 = load_afrag(&S.bufA[0][0], lane);
    v8f  d0 = {};
    v8f  d1 = {};
    d0 = wmma16x16x32(A1, bEn0, d0);
    d1 = wmma16x16x32(A1, bEn1, d1);
#pragma unroll
    for (int r = 0; r < 8; ++r) {
      S.bufB[pr0 + r][l15]      = (_Float16)fmaxf(d0[r] + b2n[l15],      0.f);
      S.bufB[pr0 + r][l15 + 16] = (_Float16)fmaxf(d1[r] + b2n[l15 + 16], 0.f);
    }
    // layer3: 32 -> 128 (8 WMMAs), fold max-over-m into registers.
    v16h A2 = load_afrag(&S.bufB[0][0], lane);
#pragma unroll
    for (int fg = 0; fg < 8; ++fg) {
      v8f e = {};
      e = wmma16x16x32(A2, bW3[fg], e);
      float mx = e[0];
#pragma unroll
      for (int r = 1; r < 8; ++r) mx = fmaxf(mx, e[r]);
      rmax[fg] = fmaxf(rmax[fg], mx);  // max over this lane's 8 m-rows
    }
  }

  // ---- cross-wave / cross-lane-half max reduction over 16 partials ----
  // Lane holds features f = fg*16 + l15; bias is max-invariant, add once.
  const int row = wave * 2 + hi;
#pragma unroll
  for (int fg = 0; fg < 8; ++fg)
    red[row][fg * 16 + l15] = rmax[fg] + b3n[fg * 16 + l15];
  __syncthreads();
  if (tid < 128) {
    float acc = red[0][tid];
#pragma unroll
    for (int r = 1; r < 16; ++r) acc = fmaxf(acc, red[r][tid]);
    out[wg * Lsz + tid] = acc;  // (B,N,L) with wg = b*N + n
  }
}

extern "C" void kernel_launch(void* const* d_in, const int* in_sizes, int n_in,
                              void* d_out, int out_size, void* d_ws,
                              size_t ws_size, hipStream_t stream) {
  const float* X     = (const float*)d_in[0];
  const float* Y     = (const float*)d_in[1];
  const float* eqw1  = (const float*)d_in[2];
  const float* eqb1  = (const float*)d_in[3];
  const float* eqw2  = (const float*)d_in[4];
  const float* eqb2  = (const float*)d_in[5];
  const float* eqw3  = (const float*)d_in[6];
  const float* eqb3  = (const float*)d_in[7];
  const float* encw1 = (const float*)d_in[8];
  const float* encb1 = (const float*)d_in[9];
  const float* encw2 = (const float*)d_in[10];
  const float* encb2 = (const float*)d_in[11];
  const float* encw3 = (const float*)d_in[12];
  const float* encb3 = (const float*)d_in[13];
  float* out = (float*)d_out;

  equirn_kernel<<<Bsz * Nsz, 256, 0, stream>>>(
      X, Y, eqw1, eqb1, eqw2, eqb2, eqw3, eqb3, encw1, encb1, encw2, encb2,
      encw3, encb3, out);
}
